// MultiHeadAttention_73340861546921
// MI455X (gfx1250) — matrix operand order repaired and checked by reading, not compiled, not run on hardware
//
#include <hip/hip_runtime.h>
#include <hip/hip_bf16.h>
#include <stdint.h>

// Problem constants (match reference)
#define HID    1024
#define NHEADS 16
#define DKV    64
#define BATCH  4
#define SEQ    2048
#define NEG_BIG (-1.0e9f)

typedef __attribute__((ext_vector_type(16))) __bf16 v16bf;
typedef __attribute__((ext_vector_type(8)))  float  v8f;

// Fragment aliases directly onto two 128-bit loads: zero repacking cost.
union Frag16 {
  v16bf v;
  uint4 q[2];
  unsigned short s[16];
};

static __device__ __forceinline__ unsigned short f2bf(float x) {
  return __builtin_bit_cast(unsigned short, (__bf16)x);   // native cvt
}

static __device__ __forceinline__ v8f wmma_bf16(const Frag16& a, const Frag16& b, v8f c) {
  return __builtin_amdgcn_wmma_f32_16x16x32_bf16(false, a.v, false, b.v, (short)0, c,
                                                 false, false);
}

// ---------------------------------------------------------------------------
// One-time conversion kernels (cheap vs the GEMMs; keep WMMA loops pure).
// ---------------------------------------------------------------------------

// fp32 -> bf16, 4 elements/thread (all sizes are multiples of 4)
__global__ void cvt4_kernel(const float* __restrict__ in,
                            unsigned short* __restrict__ out, int n4) {
  const int i = blockIdx.x * blockDim.x + threadIdx.x;
  if (i < n4) {
    const float4 f = ((const float4*)in)[i];
    union { unsigned short s[4]; uint2 u; } p;
    p.s[0] = f2bf(f.x); p.s[1] = f2bf(f.y);
    p.s[2] = f2bf(f.z); p.s[3] = f2bf(f.w);
    ((uint2*)out)[i] = p.u;
  }
}

// W [H][K=1024][N=64] fp32 -> WT [H][N=64][K=1024] bf16  (1,048,576 elems)
__global__ void wqkv_tr_kernel(const float* __restrict__ in,
                               unsigned short* __restrict__ out) {
  const int idx = blockIdx.x * blockDim.x + threadIdx.x;   // output index
  const int k = idx & 1023;
  const int n = (idx >> 10) & 63;
  const int h = idx >> 16;
  out[idx] = f2bf(in[(h << 16) + (k << 6) + n]);
}

// WO [K=1024][N=1024] fp32 -> WOT [N=1024][K=1024] bf16
__global__ void wo_tr_kernel(const float* __restrict__ in,
                             unsigned short* __restrict__ out) {
  const int idx = blockIdx.x * blockDim.x + threadIdx.x;
  const int k = idx & 1023;
  const int n = idx >> 10;
  out[idx] = f2bf(in[(k << 10) + n]);
}

// ---------------------------------------------------------------------------
// Kernel 1: per-head projection  out[16x64] = X[16x1024] * W_h[1024x64] + b_h
// One wave per (b, h, 16-row tile). Inner loop: 10x global_load_b128 + 4 WMMA.
// storeTransposed=1 writes V as [b,h,dv,s] for contiguous PV B-fragments.
// ---------------------------------------------------------------------------
__global__ void qkv_proj_kernel(const unsigned short* __restrict__ X,    // bf16 [B,S,HID]
                                const unsigned short* __restrict__ WT,   // bf16 [H,64,HID]
                                const float* __restrict__ bias,          // f32 [H,64]
                                unsigned short* __restrict__ out,
                                int storeTransposed) {
  const int lane = threadIdx.x;
  const int rlo  = lane & 15;
  const int hi   = lane >> 4;
  const int tile = blockIdx.x;                // over B*S/16
  const int b    = tile / (SEQ / 16);
  const int s0   = (tile % (SEQ / 16)) * 16;
  const int h    = blockIdx.y;

  const unsigned short* __restrict__ xrow = X + ((size_t)b * SEQ + s0 + rlo) * HID;
  const unsigned short* __restrict__ wht  = WT + ((size_t)h * DKV) * HID;

  v8f acc[4] = {};

  for (int kk = 0; kk < HID; kk += 32) {
    // A fragment: halves 0-7 -> k = kk+hi*8+i ; halves 8-15 -> +16
    Frag16 a;
    const int ka = kk + hi * 8;
    a.q[0] = *(const uint4*)(xrow + ka);
    a.q[1] = *(const uint4*)(xrow + ka + 16);

    // B fragments: lane col = rlo ; halves 0-7 -> k = kk+hi*8+i ; halves 8-15 -> +16
    const int kb = kk + hi * 8;
#pragma unroll
    for (int nc = 0; nc < 4; ++nc) {
      const unsigned short* wp = wht + (size_t)(nc * 16 + rlo) * HID + kb;
      Frag16 bf;
      bf.q[0] = *(const uint4*)(wp);
      bf.q[1] = *(const uint4*)(wp + 16);
      acc[nc] = wmma_bf16(a, bf, acc[nc]);
    }
  }

  // bias + store (C layout: lane col = rlo, VGPR r -> row r + hi*8)
#pragma unroll
  for (int nc = 0; nc < 4; ++nc) {
    const int col = nc * 16 + rlo;
    const float bb = bias[h * DKV + col];
#pragma unroll
    for (int r = 0; r < 8; ++r) {
      const int m = r + hi * 8;
      const float val = acc[nc][r] + bb;
      size_t idx;
      if (storeTransposed)
        idx = (((size_t)b * NHEADS + h) * DKV + col) * SEQ + (size_t)(s0 + m);
      else
        idx = (((size_t)b * NHEADS + h) * SEQ + (size_t)(s0 + m)) * DKV + col;
      out[idx] = f2bf(val);
    }
  }
}

// ---------------------------------------------------------------------------
// Kernel 2: flash attention. One wave per (b, h, 16-query tile).
// 32-key steps: 4 WMMAs for QK^T, masked online softmax, P transposed through
// a bf16 LDS tile (C->A layout), 4 WMMAs for P*V.
// ---------------------------------------------------------------------------
__global__ void attn_kernel(const unsigned short* __restrict__ Q,
                            const unsigned short* __restrict__ K,
                            const unsigned short* __restrict__ VT,
                            const unsigned char* __restrict__ mask,
                            unsigned short* __restrict__ outc) {
  const int lane = threadIdx.x;
  const int rlo  = lane & 15;
  const int hi   = lane >> 4;
  const int q0   = blockIdx.x * 16;
  const int h    = blockIdx.y;
  const int b    = blockIdx.z;

  // bf16 P tile, row stride 40 ushorts = 80B (16B-aligned rows, conflict-free)
  __shared__ __align__(16) unsigned short pl[16][40];

  const size_t bh = (size_t)b * NHEADS + h;

  // Q A-fragments: qa0 covers k=0..31, qa1 covers k=32..63
  const unsigned short* qrow = Q + (bh * SEQ + (size_t)(q0 + rlo)) * DKV;
  Frag16 qa0, qa1;
  qa0.q[0] = *(const uint4*)(qrow + hi * 8);
  qa0.q[1] = *(const uint4*)(qrow + hi * 8 + 16);
  qa1.q[0] = *(const uint4*)(qrow + 32 + hi * 8);
  qa1.q[1] = *(const uint4*)(qrow + 32 + hi * 8 + 16);

  v8f oacc[4] = {};
  float mrow[8], lrow[8];
#pragma unroll
  for (int r = 0; r < 8; ++r) { mrow[r] = -1.0e30f; lrow[r] = 0.0f; }

  const unsigned short* kbase  = K  + bh * SEQ * DKV;
  const unsigned short* vtbase = VT + bh * DKV * SEQ;
  const unsigned char*  mbase  = mask + ((size_t)b * SEQ + q0) * SEQ;

  for (int kt = 0; kt < SEQ; kt += 32) {
    // ---- scores: two 16x16 tiles (keys kt..kt+15, kt+16..kt+31)
    v8f c[2];
#pragma unroll
    for (int t = 0; t < 2; ++t) {
      const unsigned short* krow = kbase + (size_t)(kt + t * 16 + rlo) * DKV;
      Frag16 kb0, kb1;
      kb0.q[0] = *(const uint4*)(krow + hi * 8);
      kb0.q[1] = *(const uint4*)(krow + hi * 8 + 16);
      kb1.q[0] = *(const uint4*)(krow + 32 + hi * 8);
      kb1.q[1] = *(const uint4*)(krow + 32 + hi * 8 + 16);
      v8f cc = {};
      cc = wmma_bf16(qa0, kb0, cc);
      cc = wmma_bf16(qa1, kb1, cc);
      c[t] = cc;
    }

    // ---- scale 1/sqrt(64) + mask
#pragma unroll
    for (int t = 0; t < 2; ++t) {
#pragma unroll
      for (int r = 0; r < 8; ++r) {
        const int qr = r + hi * 8;
        const unsigned char mk = mbase[(size_t)qr * SEQ + kt + t * 16 + rlo];
        const float sc = c[t][r] * 0.125f;
        c[t][r] = mk ? sc : NEG_BIG;
      }
    }

    __syncthreads();   // previous iteration's P reads complete

    // ---- online softmax (row reductions across the 16-lane half)
#pragma unroll
    for (int r = 0; r < 8; ++r) {
      float rmax = fmaxf(c[0][r], c[1][r]);
#pragma unroll
      for (int xm = 1; xm < 16; xm <<= 1)
        rmax = fmaxf(rmax, __shfl_xor(rmax, xm, 32));
      const float mnew  = fmaxf(mrow[r], rmax);
      const float alpha = __expf(mrow[r] - mnew);
      mrow[r] = mnew;
      const float p0 = __expf(c[0][r] - mnew);
      const float p1 = __expf(c[1][r] - mnew);
      float rs = p0 + p1;
#pragma unroll
      for (int xm = 1; xm < 16; xm <<= 1)
        rs += __shfl_xor(rs, xm, 32);
      lrow[r] = lrow[r] * alpha + rs;
#pragma unroll
      for (int nc = 0; nc < 4; ++nc) oacc[nc][r] *= alpha;
      // write P (bf16) into the transpose tile: row = query, col = key
      pl[r + hi * 8][rlo]      = f2bf(p0);
      pl[r + hi * 8][16 + rlo] = f2bf(p1);
    }

    __syncthreads();

    // ---- read P back in A layout: two 128-bit LDS loads, no conversion
    Frag16 pa;
    pa.q[0] = *(const uint4*)&pl[rlo][hi * 8];
    pa.q[1] = *(const uint4*)&pl[rlo][16 + hi * 8];

    // ---- P(16x32) * V(32x64): 4 WMMAs; V pre-transposed -> contiguous loads
#pragma unroll
    for (int nc = 0; nc < 4; ++nc) {
      const unsigned short* vrow =
          vtbase + (size_t)(nc * 16 + rlo) * SEQ + kt + hi * 8;
      Frag16 vb;
      vb.q[0] = *(const uint4*)(vrow);
      vb.q[1] = *(const uint4*)(vrow + 16);
      oacc[nc] = wmma_bf16(pa, vb, oacc[nc]);
    }

    // prefetch next key tile into cache (global_prefetch_b8)
    if (kt + 32 < SEQ)
      __builtin_prefetch(kbase + (size_t)(kt + 32 + rlo) * DKV, 0, 1);
  }

  // ---- normalize and store head-concatenated output [b, s, h*64+col] as bf16
#pragma unroll
  for (int nc = 0; nc < 4; ++nc) {
#pragma unroll
    for (int r = 0; r < 8; ++r) {
      const int qr  = q0 + r + hi * 8;
      const int col = h * DKV + nc * 16 + rlo;
      const float val = oacc[nc][r] / lrow[r];
      outc[((size_t)b * SEQ + qr) * HID + col] = f2bf(val);
    }
  }
}

// ---------------------------------------------------------------------------
// Kernel 3: output projection  out[B*S,1024] = A[B*S,1024] * WO[1024,1024] + bo
// WOT is pre-transposed bf16 -> B fragments are contiguous 128-bit loads.
// ---------------------------------------------------------------------------
__global__ void out_proj_kernel(const unsigned short* __restrict__ A,
                                const unsigned short* __restrict__ WOT,  // [N][K] bf16
                                const float* __restrict__ BO,
                                float* __restrict__ out) {
  const int lane = threadIdx.x;
  const int rlo  = lane & 15;
  const int hi   = lane >> 4;
  const size_t row0 = (size_t)blockIdx.x * 16;
  const int n0 = blockIdx.y * 64;

  const unsigned short* arow = A + (row0 + rlo) * HID;
  v8f acc[4] = {};

  for (int kk = 0; kk < HID; kk += 32) {
    Frag16 a;
    const int ka = kk + hi * 8;
    a.q[0] = *(const uint4*)(arow + ka);
    a.q[1] = *(const uint4*)(arow + ka + 16);

    const int kb = kk + hi * 8;
#pragma unroll
    for (int nc = 0; nc < 4; ++nc) {
      const unsigned short* wp = WOT + (size_t)(n0 + nc * 16 + rlo) * HID + kb;
      Frag16 bf;
      bf.q[0] = *(const uint4*)(wp);
      bf.q[1] = *(const uint4*)(wp + 16);
      acc[nc] = wmma_bf16(a, bf, acc[nc]);
    }
  }

#pragma unroll
  for (int nc = 0; nc < 4; ++nc) {
    const int col = n0 + nc * 16 + rlo;
    const float bb = BO[col];
#pragma unroll
    for (int r = 0; r < 8; ++r)
      out[(row0 + (size_t)(r + hi * 8)) * HID + col] = acc[nc][r] + bb;
  }
}

// ---------------------------------------------------------------------------
extern "C" void kernel_launch(void* const* d_in, const int* in_sizes, int n_in,
                              void* d_out, int out_size, void* d_ws, size_t ws_size,
                              hipStream_t stream) {
  (void)in_sizes; (void)n_in; (void)out_size; (void)ws_size;

  const float*         xq   = (const float*)d_in[0];
  const float*         xk   = (const float*)d_in[1];
  const float*         xv   = (const float*)d_in[2];
  const unsigned char* mask = (const unsigned char*)d_in[3];
  const float*         wq   = (const float*)d_in[4];
  const float*         bq   = (const float*)d_in[5];
  const float*         wk   = (const float*)d_in[6];
  const float*         bk   = (const float*)d_in[7];
  const float*         wv   = (const float*)d_in[8];
  const float*         bv   = (const float*)d_in[9];
  const float*         wo   = (const float*)d_in[10];
  const float*         bo   = (const float*)d_in[11];
  float*               out  = (float*)d_out;

  // Workspace (bf16 ushort elements):
  //  xq/xk/xv bf16  : 3 x B*S*HID        = 3 x 8,388,608
  //  wqt/wkt/wvt    : 3 x H*64*HID       = 3 x 1,048,576
  //  wot            : 1 x HID*HID        = 1,048,576
  //  q/k/vt buffers : 3 x B*H*S*64       = 3 x 8,388,608
  //  attn concat    : aliases xq-bf16 buffer (stream-ordered reuse)
  unsigned short* ws = (unsigned short*)d_ws;
  const size_t XE = (size_t)BATCH * SEQ * HID;         // 8,388,608
  const size_t WE = (size_t)NHEADS * DKV * HID;        // 1,048,576
  unsigned short* xqbf = ws;
  unsigned short* xkbf = xqbf + XE;
  unsigned short* xvbf = xkbf + XE;
  unsigned short* wqt  = xvbf + XE;
  unsigned short* wkt  = wqt + WE;
  unsigned short* wvt  = wkt + WE;
  unsigned short* wot  = wvt + WE;
  unsigned short* qbuf = wot + WE;
  unsigned short* kbuf = qbuf + XE;
  unsigned short* vtbf = kbuf + XE;
  unsigned short* abuf = xqbf;   // reuse: xq-bf16 dead after qkv_proj(q)

  dim3 blk256(256), blk32(32);

  // one-time conversions / transposes
  const int n4 = (int)(XE / 4);
  hipLaunchKernelGGL(cvt4_kernel, dim3((n4 + 255) / 256), blk256, 0, stream, xq, xqbf, n4);
  hipLaunchKernelGGL(cvt4_kernel, dim3((n4 + 255) / 256), blk256, 0, stream, xk, xkbf, n4);
  hipLaunchKernelGGL(cvt4_kernel, dim3((n4 + 255) / 256), blk256, 0, stream, xv, xvbf, n4);
  hipLaunchKernelGGL(wqkv_tr_kernel, dim3(WE / 256), blk256, 0, stream, wq, wqt);
  hipLaunchKernelGGL(wqkv_tr_kernel, dim3(WE / 256), blk256, 0, stream, wk, wkt);
  hipLaunchKernelGGL(wqkv_tr_kernel, dim3(WE / 256), blk256, 0, stream, wv, wvt);
  hipLaunchKernelGGL(wo_tr_kernel,   dim3(WE / 256), blk256, 0, stream, wo, wot);

  // projections
  dim3 gproj(BATCH * SEQ / 16, NHEADS);
  hipLaunchKernelGGL(qkv_proj_kernel, gproj, blk32, 0, stream, xqbf, wqt, bq, qbuf, 0);
  hipLaunchKernelGGL(qkv_proj_kernel, gproj, blk32, 0, stream, xkbf, wkt, bk, kbuf, 0);
  hipLaunchKernelGGL(qkv_proj_kernel, gproj, blk32, 0, stream, xvbf, wvt, bv, vtbf, 1);

  // attention
  dim3 gattn(SEQ / 16, NHEADS, BATCH);
  hipLaunchKernelGGL(attn_kernel, gattn, blk32, 0, stream, qbuf, kbuf, vtbf, mask, abuf);

  // output projection
  dim3 gout(BATCH * SEQ / 16, HID / 64);
  hipLaunchKernelGGL(out_proj_kernel, gout, blk32, 0, stream, abuf, wot, bo, out);
}
